// Encoder_755914244431
// MI455X (gfx1250) — matrix operand order repaired and checked by reading, not compiled, not run on hardware
//
#include <hip/hip_runtime.h>
#include <cstdint>
#include <cstddef>

#define LSTM_B 128
#define LSTM_T 512
#define LSTM_F 256
#define LSTM_H 512
#define LSTM_K (LSTM_F + LSTM_H)   // 768 = fused [x | h] contraction dim
#define LSTM_4H (4 * LSTM_H)       // 2048

typedef __bf16 v16bf __attribute__((ext_vector_type(16)));
typedef __bf16 v8bf  __attribute__((ext_vector_type(8)));
typedef __bf16 v4bf  __attribute__((ext_vector_type(4)));
typedef float  v8f   __attribute__((ext_vector_type(8)));

__device__ __forceinline__ float sigmoid_f(float x) {
    return 1.0f / (1.0f + __expf(-x));
}

// ---------------------------------------------------------------------------
// Pre-pass: f32 -> bf16 conversion of x (vectorized, one-time, ~64MB read)
// ---------------------------------------------------------------------------
__global__ __launch_bounds__(256) void convert_x_kernel(
    const float* __restrict__ src, __bf16* __restrict__ dst, int n4)
{
    int i = blockIdx.x * blockDim.x + threadIdx.x;
    if (i >= n4) return;
    float4 v = ((const float4*)src)[i];
    v4bf o;
    o[0] = (__bf16)v.x; o[1] = (__bf16)v.y; o[2] = (__bf16)v.z; o[3] = (__bf16)v.w;
    ((v4bf*)dst)[i] = o;
}

// ---------------------------------------------------------------------------
// Pre-pass: build fused, transposed bf16 weights Wt[n][k], n in [0,4H), k in
// [0,768): k<256 -> kernel[k][n], else recurrent_kernel[k-256][n].
// Transposed layout makes the WMMA B-operand load (per-lane: fixed N column,
// 16 contiguous K values) a contiguous 32-byte load.
// ---------------------------------------------------------------------------
__global__ __launch_bounds__(256) void build_wt_kernel(
    const float* __restrict__ kern, const float* __restrict__ rec,
    __bf16* __restrict__ Wt)
{
    int idx = blockIdx.x * blockDim.x + threadIdx.x;  // 0 .. 4H*K-1
    int k = idx % LSTM_K;
    int n = idx / LSTM_K;
    float v = (k < LSTM_F) ? kern[(size_t)k * LSTM_4H + n]
                           : rec[(size_t)(k - LSTM_F) * LSTM_4H + n];
    Wt[idx] = (__bf16)v;
}

// ---------------------------------------------------------------------------
// Pre-pass: zero initial c and h (ws is poisoned 0xAA by the harness)
// ---------------------------------------------------------------------------
__global__ __launch_bounds__(256) void init_state_kernel(
    float* __restrict__ c, __bf16* __restrict__ h0, __bf16* __restrict__ h1)
{
    int i = blockIdx.x * blockDim.x + threadIdx.x;
    if (i < LSTM_B * LSTM_H) {
        c[i]  = 0.0f;
        h0[i] = (__bf16)0.0f;
        h1[i] = (__bf16)0.0f;
    }
}

// ---------------------------------------------------------------------------
// One LSTM timestep, fully fused:  z = [x_t | h_{t-1}] @ Wt^T + bias, gates,
// cell update, write h_t (bf16 state + f32 output slice) and c (in place).
//
// Tiling: grid = (4, 8) blocks x 256 threads (8 waves).
//   blockIdx.y       = mtile   (16 batch rows, 8 tiles cover B=128)
//   blockIdx.x*8+wave= ngroup  (16 hidden cols, 32 groups cover H=512)
// Each wave computes the SAME 16 hidden columns in all four gate blocks
// (cols n0+g*512), so the pointwise cell update needs no cross-wave exchange.
// K loop: 24 slabs of 32 (8 from x_t, 16 from h_{t-1}); 4 WMMAs per slab.
// ---------------------------------------------------------------------------
__global__ __launch_bounds__(256) void lstm_step_kernel(
    const __bf16* __restrict__ xb,    // [B,T,F] bf16
    const __bf16* __restrict__ Wt,    // [4H, 768] bf16 (transposed weights)
    const float*  __restrict__ bias,  // [4H] f32
    const __bf16* __restrict__ h_in,  // [B,H] bf16
    __bf16*       __restrict__ h_out, // [B,H] bf16
    float*        __restrict__ c_st,  // [B,H] f32, updated in place
    float*        __restrict__ out,   // [B,T,H] f32
    int t)
{
    const int tid  = threadIdx.x;
    const int wave = tid >> 5;
    const int lane = tid & 31;
    const int half = lane >> 4;   // 0 or 1 (which 16-lane half)
    const int l16  = lane & 15;

    const int mtile  = blockIdx.y;                 // 0..7
    const int ngroup = blockIdx.x * 8 + wave;      // 0..31
    const int n0     = ngroup * 16;                // hidden-col base per gate
    const int mbase  = mtile * 16;

    // A operand (16x32 bf16, MxK): lane holds row M=l16; its 16 elements are
    // K = half*16 + {0..15}  (same contiguous order as B).
    const int arow = mbase + l16;                  // batch row
    const __bf16* xrow = xb  + ((size_t)arow * LSTM_T + t) * LSTM_F;
    const __bf16* hrow = h_in + (size_t)arow * LSTM_H;
    const int ka = half * 8;   // unused; A loads use kb, same as B
    const int kb = half * 16;  // B chunk offset inside a 32-wide K slab

    // B operand (32x16 bf16, KxN): lane holds column N=l16; its 16 elements
    // are K = half*16 + {0..15}, contiguous in the transposed Wt row.
    const __bf16* wrow0 = Wt + (size_t)(0 * LSTM_H + n0 + l16) * LSTM_K;
    const __bf16* wrow1 = Wt + (size_t)(1 * LSTM_H + n0 + l16) * LSTM_K;
    const __bf16* wrow2 = Wt + (size_t)(2 * LSTM_H + n0 + l16) * LSTM_K;
    const __bf16* wrow3 = Wt + (size_t)(3 * LSTM_H + n0 + l16) * LSTM_K;

    v8f acc0 = {}, acc1 = {}, acc2 = {}, acc3 = {};

    // ---- K slabs 0..7: contraction over x_t (F = 256) ----
#pragma unroll
    for (int ks = 0; ks < LSTM_F / 32; ++ks) {
        const int k32 = ks * 32;
        v16bf a  = *(const v16bf*)(xrow + k32 + kb);
        v16bf b0 = *(const v16bf*)(wrow0 + k32 + kb);
        v16bf b1 = *(const v16bf*)(wrow1 + k32 + kb);
        v16bf b2 = *(const v16bf*)(wrow2 + k32 + kb);
        v16bf b3 = *(const v16bf*)(wrow3 + k32 + kb);
        acc0 = __builtin_amdgcn_wmma_f32_16x16x32_bf16(false, a, false, b0, (short)0, acc0, false, false);
        acc1 = __builtin_amdgcn_wmma_f32_16x16x32_bf16(false, a, false, b1, (short)0, acc1, false, false);
        acc2 = __builtin_amdgcn_wmma_f32_16x16x32_bf16(false, a, false, b2, (short)0, acc2, false, false);
        acc3 = __builtin_amdgcn_wmma_f32_16x16x32_bf16(false, a, false, b3, (short)0, acc3, false, false);
    }

    // ---- K slabs 8..23: contraction over h_{t-1} (H = 512) ----
#pragma unroll 4
    for (int ks = 0; ks < LSTM_H / 32; ++ks) {
        const int k32 = ks * 32;
        v16bf a  = *(const v16bf*)(hrow + k32 + kb);
        const int kw = LSTM_F + k32 + kb;
        v16bf b0 = *(const v16bf*)(wrow0 + kw);
        v16bf b1 = *(const v16bf*)(wrow1 + kw);
        v16bf b2 = *(const v16bf*)(wrow2 + kw);
        v16bf b3 = *(const v16bf*)(wrow3 + kw);
        acc0 = __builtin_amdgcn_wmma_f32_16x16x32_bf16(false, a, false, b0, (short)0, acc0, false, false);
        acc1 = __builtin_amdgcn_wmma_f32_16x16x32_bf16(false, a, false, b1, (short)0, acc1, false, false);
        acc2 = __builtin_amdgcn_wmma_f32_16x16x32_bf16(false, a, false, b2, (short)0, acc2, false, false);
        acc3 = __builtin_amdgcn_wmma_f32_16x16x32_bf16(false, a, false, b3, (short)0, acc3, false, false);
    }

    // ---- Pointwise LSTM cell update (f32). C/D layout: element e of the
    // v8f at lane L is (M = e + 8*half, N = l16). Bias depends on N only.
    const int   colh = n0 + l16;
    const float bi = bias[0 * LSTM_H + colh];
    const float bf = bias[1 * LSTM_H + colh];
    const float bg = bias[2 * LSTM_H + colh];
    const float bo = bias[3 * LSTM_H + colh];

#pragma unroll
    for (int e = 0; e < 8; ++e) {
        const int m = mbase + half * 8 + e;     // batch row of this element
        const size_t sidx = (size_t)m * LSTM_H + colh;
        float ig = sigmoid_f(acc0[e] + bi);
        float fg = sigmoid_f(acc1[e] + bf);
        float gg = tanhf(acc2[e] + bg);
        float og = sigmoid_f(acc3[e] + bo);
        float c_new = fg * c_st[sidx] + ig * gg;
        float h_new = og * tanhf(c_new);
        c_st[sidx]  = c_new;
        h_out[sidx] = (__bf16)h_new;
        out[((size_t)m * LSTM_T + t) * LSTM_H + colh] = h_new;
    }
}

// ---------------------------------------------------------------------------
// Epilogue: h_last = output[:, T-1, :] (exact f32), c_last = c state
// ---------------------------------------------------------------------------
__global__ __launch_bounds__(256) void finalize_kernel(
    const float* __restrict__ out, const float* __restrict__ c,
    float* __restrict__ hlast, float* __restrict__ clast)
{
    int i = blockIdx.x * blockDim.x + threadIdx.x;
    if (i < LSTM_B * LSTM_H) {
        int b = i / LSTM_H, h = i % LSTM_H;
        hlast[i] = out[((size_t)b * LSTM_T + (LSTM_T - 1)) * LSTM_H + h];
        clast[i] = c[i];
    }
}

// ---------------------------------------------------------------------------
extern "C" void kernel_launch(void* const* d_in, const int* in_sizes, int n_in,
                              void* d_out, int out_size, void* d_ws, size_t ws_size,
                              hipStream_t stream) {
    (void)in_sizes; (void)n_in; (void)out_size; (void)ws_size;

    const float* x    = (const float*)d_in[0];  // [B,T,F]
    const float* kern = (const float*)d_in[1];  // [F,4H]
    const float* rec  = (const float*)d_in[2];  // [H,4H]
    const float* bias = (const float*)d_in[3];  // [4H]

    float* out   = (float*)d_out;                                  // [B,T,H]
    float* hlast = out + (size_t)LSTM_B * LSTM_T * LSTM_H;         // [B,H]
    float* clast = hlast + (size_t)LSTM_B * LSTM_H;                // [B,H]

    // Workspace layout (~36 MB total)
    char* ws = (char*)d_ws;
    const size_t wt_bytes = (size_t)LSTM_4H * LSTM_K * sizeof(__bf16);            // 3 MB
    const size_t xb_bytes = (size_t)LSTM_B * LSTM_T * LSTM_F * sizeof(__bf16);    // 32 MB
    const size_t c_bytes  = (size_t)LSTM_B * LSTM_H * sizeof(float);              // 256 KB
    const size_t h_bytes  = (size_t)LSTM_B * LSTM_H * sizeof(__bf16);             // 128 KB
    __bf16* Wt = (__bf16*)(ws);
    __bf16* xb = (__bf16*)(ws + wt_bytes);
    float*  cs = (float*) (ws + wt_bytes + xb_bytes);
    __bf16* h0 = (__bf16*)(ws + wt_bytes + xb_bytes + c_bytes);
    __bf16* h1 = (__bf16*)(ws + wt_bytes + xb_bytes + c_bytes + h_bytes);

    // One-time pre-passes
    {
        int n4 = LSTM_B * LSTM_T * LSTM_F / 4;
        convert_x_kernel<<<n4 / 256, 256, 0, stream>>>(x, xb, n4);
    }
    build_wt_kernel<<<(LSTM_4H * LSTM_K) / 256, 256, 0, stream>>>(kern, rec, Wt);
    init_state_kernel<<<(LSTM_B * LSTM_H) / 256, 256, 0, stream>>>(cs, h0, h1);

    // Sequential scan: one fused kernel per timestep, ping-pong h buffers.
    for (int t = 0; t < LSTM_T; ++t) {
        const __bf16* hin  = (t & 1) ? h1 : h0;
        __bf16*       hout = (t & 1) ? h0 : h1;
        lstm_step_kernel<<<dim3(4, 8), 256, 0, stream>>>(
            xb, Wt, bias, hin, hout, cs, out, t);
    }

    finalize_kernel<<<(LSTM_B * LSTM_H) / 256, 256, 0, stream>>>(out, cs, hlast, clast);
}
